// Attention_4990751998614
// MI455X (gfx1250) — matrix operand order repaired and checked by reading, not compiled, not run on hardware
//
#include <hip/hip_runtime.h>

// Problem constants (B=1)
#define QN 2048
#define KN 2048
#define CIN 256
#define NH 8
#define CHD 32
#define DMODEL 256   // NH*CHD

typedef __attribute__((ext_vector_type(16))) __bf16 v16bf;
typedef __attribute__((ext_vector_type(8)))  float  v8f;
typedef __attribute__((ext_vector_type(2)))  float  v2f;
typedef __attribute__((ext_vector_type(4)))  unsigned int u32x4;
typedef __attribute__((ext_vector_type(8)))  int    i32x8;
typedef __attribute__((ext_vector_type(4)))  int    i32x4;

#if defined(__HIP_DEVICE_COMPILE__)
#define HAS_WMMA_F32 __has_builtin(__builtin_amdgcn_wmma_f32_16x16x4_f32)
#else
#define HAS_WMMA_F32 0
#endif

// ---------------------------------------------------------------------------
// Tensor Data Mover: async copy of a 16-row x 32-col f32 tile (row stride KN)
// from global memory into LDS at byte offset lds_off.  Descriptor bitfields
// per CDNA5 ISA 8.3-8.6 (group0: count/lds/global/type, group1: dims/strides).
// ---------------------------------------------------------------------------
__device__ __forceinline__ void tdm_load_pb_tile(unsigned lds_off, const float* gaddr) {
    unsigned long long ga = (unsigned long long)(uintptr_t)gaddr;
    unsigned ga_lo = (unsigned)(ga & 0xffffffffu);
    unsigned ga_hi = (unsigned)((ga >> 32) & 0x1ffffffu);

    u32x4 g0;
    g0[0] = 1u;                                                    // count=1 (valid), user mode
    g0[1] = (unsigned)__builtin_amdgcn_readfirstlane((int)lds_off); // lds_addr
    g0[2] = (unsigned)__builtin_amdgcn_readfirstlane((int)ga_lo);   // global_addr[31:0]
    g0[3] = (unsigned)__builtin_amdgcn_readfirstlane((int)ga_hi)
            | 0x80000000u;                                          // addr[56:32] | type=2

    i32x8 g1;
    g1[0] = (int)(2u << 16);                       // data_size = 4 bytes
    g1[1] = (int)((unsigned)(KN & 0xffff) << 16);  // tensor_dim0[15:0]
    g1[2] = (int)(((unsigned)KN >> 16) |
                  ((unsigned)(QN & 0xffff) << 16)); // tensor_dim0[31:16] | tensor_dim1[15:0]
    g1[3] = (int)((32u) << 16);                    // tensor_dim1[31:16]=0 | tile_dim0=32
    g1[4] = 16;                                    // tile_dim1=16, tile_dim2=0
    g1[5] = KN;                                    // tensor_dim0_stride[31:0]
    g1[6] = 0;                                     // stride0[47:32]=0, dim1_stride[15:0]=0
    g1[7] = 0;

    i32x4 g2 = {};                                 // 2D tile: dims 2/3 unused
    i32x4 g3 = {};
#if __clang_major__ >= 23
    i32x8 g4 = {};
    __builtin_amdgcn_tensor_load_to_lds(g0, g1, g2, g3, g4, 0);
#else
    __builtin_amdgcn_tensor_load_to_lds(g0, g1, g2, g3, 0);
#endif
}

// ---------------------------------------------------------------------------
// 16x16 f32 output tile GEMM: C = X[row0:+16, 0:kdim] * W[0:kdim, col0:+16]
// ---------------------------------------------------------------------------
__device__ __forceinline__ v8f gemm_tile_16x16(const float* __restrict__ X,
                                               const float* __restrict__ W,
                                               int ldx, int ldw,
                                               int row0, int col0, int kdim,
                                               int lane) {
    const int half = lane >> 4;
    const int r    = lane & 15;
    v8f acc = {};
#if HAS_WMMA_F32
    for (int kk = 0; kk < kdim; kk += 4) {
        v2f a, b;
#pragma unroll
        for (int j = 0; j < 2; ++j) {
            a[j] = X[(size_t)(row0 + r) * ldx + kk + 2 * half + j];   // A: M=r, K=j+2*half
            b[j] = W[(size_t)(kk + 2 * half + j) * ldw + col0 + r];   // B: N=r, K=j+2*half
        }
        acc = __builtin_amdgcn_wmma_f32_16x16x4_f32(false, a, false, b,
                                                    (short)0, acc, false, false);
    }
#else
    for (int kk = 0; kk < kdim; kk += 32) {
        v16bf a, b;
#pragma unroll
        for (int j = 0; j < 16; ++j) {
            const int ka = kk + j + 16 * half;
            const int kb = kk + j + 16 * half;
            a[j] = (__bf16)X[(size_t)(row0 + r) * ldx + ka];
            b[j] = (__bf16)W[(size_t)kb * ldw + col0 + r];
        }
        acc = __builtin_amdgcn_wmma_f32_16x16x32_bf16(false, a, false, b,
                                                      (short)0, acc, false, false);
    }
#endif
    return acc;
}

// ---------------------------------------------------------------------------
// Kernel 1: fused Q/K/V/G projections.  grid=(256,4), block=256 (8 waves)
// ---------------------------------------------------------------------------
__global__ __launch_bounds__(256)
void proj_kernel(const float* __restrict__ q_x, const float* __restrict__ kv_x,
                 const float* __restrict__ w_q, const float* __restrict__ w_k,
                 const float* __restrict__ w_v, const float* __restrict__ w_g,
                 const float* __restrict__ b_g,
                 __bf16* __restrict__ qb, __bf16* __restrict__ kb,
                 __bf16* __restrict__ vb, float* __restrict__ gb) {
    const int mode = blockIdx.y;
    const int wave = blockIdx.x * (blockDim.x >> 5) + (threadIdx.x >> 5);
    const int lane = threadIdx.x & 31;
    const int half = lane >> 4;
    const int r    = lane & 15;
    const int row0 = (wave >> 4) << 4;
    const int col0 = (wave & 15) << 4;

    const float* X = (mode == 1 || mode == 2) ? kv_x : q_x;
    const float* W = (mode == 0) ? w_q : (mode == 1) ? w_k : (mode == 2) ? w_v : w_g;

    v8f acc = gemm_tile_16x16(X, W, CIN, DMODEL, row0, col0, CIN, lane);

    const float qscale = 0.17677669529663687f; // 1/sqrt(32)
#pragma unroll
    for (int i = 0; i < 8; ++i) {
        const int row  = row0 + i + 8 * half;
        const int col  = col0 + r;
        const int head = col >> 5;
        const int ch   = col & 31;
        const size_t hidx = ((size_t)head * QN + row) * CHD + ch;
        const float v = acc[i];
        if (mode == 0)      qb[hidx] = (__bf16)(v * qscale);
        else if (mode == 1) kb[hidx] = (__bf16)v;
        else if (mode == 2) vb[hidx] = (__bf16)v;
        else                gb[(size_t)row * DMODEL + col] =
                                1.0f / (1.0f + __expf(-(v + b_g[col])));
    }
}

// ---------------------------------------------------------------------------
// Kernel 2: fused flash attention with pair bias + gating.
// One wave per (head, 16-query tile); 32-key steps.  pair_bias streamed by the
// Tensor Data Mover into double-buffered LDS tiles (TENSORcnt-tracked).
// LDS: per wave 2KB P-bounce + 2x2KB pair tiles; 8 waves -> 48KB.
// ---------------------------------------------------------------------------
__global__ __launch_bounds__(256)
void attn_kernel(const __bf16* __restrict__ qb, const __bf16* __restrict__ kbuf,
                 const __bf16* __restrict__ vbuf,
                 const float* __restrict__ pb, const float* __restrict__ mb,
                 const float* __restrict__ gb, float* __restrict__ ob) {
    __shared__ __align__(16) char smem[16384 + 32768];

    const int wslot = threadIdx.x >> 5;
    const int lane  = threadIdx.x & 31;
    const int half  = lane >> 4;
    const int r     = lane & 15;
    const int wsl   = __builtin_amdgcn_readfirstlane(wslot);   // uniform wave slot
    const int wave  = blockIdx.x * 8 + wsl;
    const int h     = wave >> 7;           // / (QN/16)
    const int q0    = (wave & 127) << 4;

    float* ldsw = (float*)(smem + wslot * 2048);               // P bounce [16x32 f32]
    const unsigned pbt_off = 16384u + (unsigned)wsl * 4096u;   // pair-tile double buffer

    // Q tile as WMMA A operand (held for the whole K loop)
    v16bf aq;
    {
        const __bf16* qrow = qb + ((size_t)h * QN + q0 + r) * CHD;
#pragma unroll
        for (int j = 0; j < 16; ++j) {
            const int ch = (j < 8) ? (j + 8 * half) : (j + 8 + 8 * half);
            aq[j] = qrow[ch];
        }
    }

    float m_run[8], l_run[8];
    v8f o0 = {}, o1 = {};
#pragma unroll
    for (int i = 0; i < 8; ++i) { m_run[i] = -3.0e38f; l_run[i] = 0.0f; }

    const float* pbase = pb + ((size_t)h * QN + q0) * KN;

    // prime double buffer: tile for k0=0 into buffer 0
    tdm_load_pb_tile(pbt_off, pbase);

    for (int k0 = 0; k0 < KN; k0 += 32) {
        const int buf = (k0 >> 5) & 1;

        // issue DMA for next tile while we work on this one
        if (k0 + 32 < KN) {
            tdm_load_pb_tile(pbt_off + (unsigned)((buf ^ 1) * 2048), pbase + k0 + 32);
            __builtin_amdgcn_s_wait_tensorcnt(1);   // current tile complete
        } else {
            __builtin_amdgcn_s_wait_tensorcnt(0);
        }
        asm volatile("" ::: "memory");              // keep LDS reads below the wait
        const float* ptl = (const float*)(smem + pbt_off + (unsigned)buf * 2048);

        // ---- S = Q * K^T for two 16-key sub-tiles
        v16bf b0, b1;
        {
            const __bf16* krow0 = kbuf + ((size_t)h * KN + k0 + r) * CHD;
            const __bf16* krow1 = krow0 + 16 * CHD;
#pragma unroll
            for (int j = 0; j < 16; ++j) {
                const int ch = (j < 8) ? (j + 8 * half) : (j + 8 + 8 * half);
                b0[j] = krow0[ch];
                b1[j] = krow1[ch];
            }
        }
        v8f s0 = {}, s1 = {};
        s0 = __builtin_amdgcn_wmma_f32_16x16x32_bf16(false, aq, false, b0,
                                                     (short)0, s0, false, false);
        s1 = __builtin_amdgcn_wmma_f32_16x16x32_bf16(false, aq, false, b1,
                                                     (short)0, s1, false, false);

        const float mb0 = mb[k0 + r];
        const float mb1 = mb[k0 + 16 + r];

        // ---- online softmax; per-lane partial row sums (reduced once at end)
#pragma unroll
        for (int i = 0; i < 8; ++i) {
            const int mrow = i + 8 * half;
            float e0 = s0[i] + ptl[mrow * 32 + r]      + mb0;
            float e1 = s1[i] + ptl[mrow * 32 + 16 + r] + mb1;

            float t = fmaxf(e0, e1);
#pragma unroll
            for (int off = 1; off < 16; off <<= 1)
                t = fmaxf(t, __shfl_xor(t, off, 32));
            const float newm  = fmaxf(m_run[i], t);
            const float scale = __expf(m_run[i] - newm);   // row-uniform
            const float p0 = __expf(e0 - newm);
            const float p1 = __expf(e1 - newm);

            l_run[i] = l_run[i] * scale + (p0 + p1);       // per-lane partial
            m_run[i] = newm;
            o0[i] *= scale;
            o1[i] *= scale;

            ldsw[mrow * 32 + r]      = p0;
            ldsw[mrow * 32 + 16 + r] = p1;
        }
        asm volatile("s_wait_dscnt 0" ::: "memory");

        // ---- P (C-layout in LDS) -> A-layout bf16 operand
        v16bf ap;
#pragma unroll
        for (int j = 0; j < 16; ++j) {
            const int kl = j + 16 * half;
            ap[j] = (__bf16)ldsw[r * 32 + kl];
        }

        // ---- V tile as B operands (ch 0-15 / 16-31)
        v16bf bv0, bv1;
#pragma unroll
        for (int j = 0; j < 16; ++j) {
            const int key = k0 + j + 16 * half;
            const __bf16* vrow = vbuf + ((size_t)h * KN + key) * CHD;
            bv0[j] = vrow[r];
            bv1[j] = vrow[16 + r];
        }
        o0 = __builtin_amdgcn_wmma_f32_16x16x32_bf16(false, ap, false, bv0,
                                                     (short)0, o0, false, false);
        o1 = __builtin_amdgcn_wmma_f32_16x16x32_bf16(false, ap, false, bv1,
                                                     (short)0, o1, false, false);
    }

    // ---- epilogue: reduce per-lane row sums, normalize, sigmoid gate
#pragma unroll
    for (int i = 0; i < 8; ++i) {
        float l = l_run[i];
#pragma unroll
        for (int off = 1; off < 16; off <<= 1)
            l += __shfl_xor(l, off, 32);
        const int   mrow = i + 8 * half;
        const float inv  = 1.0f / l;
        const size_t base = (size_t)(q0 + mrow) * DMODEL + h * CHD;
        ob[base + r]      = o0[i] * inv * gb[base + r];
        ob[base + 16 + r] = o1[i] * inv * gb[base + 16 + r];
    }
}

// ---------------------------------------------------------------------------
// Kernel 3: output projection  out = O_gated * w_o + b_o.  grid=256, block=256
// ---------------------------------------------------------------------------
__global__ __launch_bounds__(256)
void oproj_kernel(const float* __restrict__ ob, const float* __restrict__ w_o,
                  const float* __restrict__ b_o, float* __restrict__ out) {
    const int wave = blockIdx.x * (blockDim.x >> 5) + (threadIdx.x >> 5);
    const int lane = threadIdx.x & 31;
    const int half = lane >> 4;
    const int r    = lane & 15;
    const int row0 = (wave >> 4) << 4;
    const int col0 = (wave & 15) << 4;

    v8f acc = gemm_tile_16x16(ob, w_o, DMODEL, CIN, row0, col0, DMODEL, lane);

#pragma unroll
    for (int i = 0; i < 8; ++i) {
        const int row = row0 + i + 8 * half;
        const int col = col0 + r;
        out[(size_t)row * CIN + col] = acc[i] + b_o[col];
    }
}

// ---------------------------------------------------------------------------
extern "C" void kernel_launch(void* const* d_in, const int* in_sizes, int n_in,
                              void* d_out, int out_size, void* d_ws, size_t ws_size,
                              hipStream_t stream) {
    (void)in_sizes; (void)n_in; (void)out_size; (void)ws_size;
    const float* q_x       = (const float*)d_in[0];
    const float* kv_x      = (const float*)d_in[1];
    const float* pair_bias = (const float*)d_in[2];
    const float* mask_bias = (const float*)d_in[3];
    const float* w_q       = (const float*)d_in[4];
    const float* w_k       = (const float*)d_in[5];
    const float* w_v       = (const float*)d_in[6];
    const float* w_g       = (const float*)d_in[7];
    const float* b_g       = (const float*)d_in[8];
    const float* w_o       = (const float*)d_in[9];
    const float* b_o       = (const float*)d_in[10];
    float* out = (float*)d_out;

    // workspace carve-up (~7 MB)
    char* ws = (char*)d_ws;
    __bf16* qb   = (__bf16*)ws;  ws += (size_t)NH * QN * CHD * sizeof(__bf16);
    __bf16* kbuf = (__bf16*)ws;  ws += (size_t)NH * KN * CHD * sizeof(__bf16);
    __bf16* vbuf = (__bf16*)ws;  ws += (size_t)NH * KN * CHD * sizeof(__bf16);
    float*  gb   = (float*)ws;   ws += (size_t)QN * DMODEL * sizeof(float);
    float*  ob   = (float*)ws;

    proj_kernel<<<dim3(256, 4), 256, 0, stream>>>(q_x, kv_x, w_q, w_k, w_v, w_g,
                                                  b_g, qb, kbuf, vbuf, gb);
    attn_kernel<<<dim3(128), 256, 0, stream>>>(qb, kbuf, vbuf, pair_bias,
                                               mask_bias, gb, ob);
    oproj_kernel<<<dim3(256), 256, 0, stream>>>(ob, w_o, b_o, out);
}
